// PlaceGNN_56959856280353
// MI455X (gfx1250) — compile-verified, matrix operand order checked by reading, not run on hardware
//
#include <hip/hip_runtime.h>
#include <hip/hip_bf16.h>
#include <math.h>

typedef __attribute__((ext_vector_type(16))) _Float16     v16h;
typedef __attribute__((ext_vector_type(8)))  float        v8f;
typedef __attribute__((ext_vector_type(4)))  unsigned int u32x4;
typedef __attribute__((ext_vector_type(8)))  int          i32x8;
typedef __attribute__((ext_vector_type(4)))  int          i32x4;

#define HID   64
#define PHID  16
#define CIN   16
#define PINF  8
#define NHEAD 4
#define HD    16
#define DMAX  8
#define KQP   96     // 80 padded to 3 k-tiles of 32
#define NQ    192    // q|k|v columns

#if defined(__gfx1250__)
#if __has_builtin(__builtin_amdgcn_tensor_load_to_lds)
#define USE_TDM 1
#endif
#endif

#ifdef USE_TDM
// DMA a contiguous byte range (8B-aligned, multiple of 8B, bytes/8 < 65536)
// from global memory into LDS via the Tensor Data Mover. Issue from ONE wave.
__device__ __forceinline__ void tdm_copy_to_lds(const void* gsrc, void* ldst,
                                                unsigned nbytes) {
  unsigned long long ga = (unsigned long long)(uintptr_t)gsrc;
  unsigned lds = (unsigned)(uintptr_t)ldst;  // low 32 bits = LDS byte offset
  unsigned n8 = nbytes >> 3;                 // 8-byte elements

  u32x4 g0;
  g0[0] = 1u;                                           // count=1 (valid user D#)
  g0[1] = lds;                                          // lds_addr
  g0[2] = (unsigned)(ga & 0xffffffffu);                 // global_addr[31:0]
  g0[3] = (unsigned)((ga >> 32) & 0x1ffffffu) | (2u << 30);  // addr[56:32] | type=2

  i32x8 g1;
  g1[0] = (int)(3u << 16);                   // wg_mask=0, data_size=3 (8B)
  g1[1] = (int)((n8 & 0xffffu) << 16);       // tensor_dim0[15:0]
  g1[2] = (int)(((n8 >> 16) & 0xffffu) | (1u << 16));  // td0[31:16] | tensor_dim1=1
  g1[3] = (int)((n8 & 0xffffu) << 16);       // tile_dim0 = n8 (fits 16 bits)
  g1[4] = 1;                                 // tile_dim1=1, tile_dim2=0
  g1[5] = (int)n8;                           // tensor_dim0_stride[31:0]
  g1[6] = 0;
  g1[7] = 0;

  i32x4 z4 = {0, 0, 0, 0};
#if __clang_major__ >= 23
  i32x8 z8 = {0, 0, 0, 0, 0, 0, 0, 0};
  __builtin_amdgcn_tensor_load_to_lds(g0, g1, z4, z4, z8, 0);
#else
  __builtin_amdgcn_tensor_load_to_lds(g0, g1, z4, z4, 0);
#endif
  __builtin_amdgcn_s_wait_tensorcnt(0);
}
#endif

// ---------------------------------------------------------------- utilities
__global__ __launch_bounds__(256) void zero_f32(float* __restrict__ p, long n) {
  long i = (long)blockIdx.x * blockDim.x + threadIdx.x;
  if (i < n) p[i] = 0.f;
}

// -------- pre-pack all layer weights into f16 images matching the LDS tiles
// wqkvh: [L][KQP][NQ]   (rows >=80 zero; cols 0..63=Wq, 64..127=Wk, 128..191=Wv)
// wsageh:[L][128][HID]  (rows 0..63=Wself, 64..127=Wneigh)
__global__ __launch_bounds__(256) void pack_weights_f16(
    const float* __restrict__ Wq, const float* __restrict__ Wk,
    const float* __restrict__ Wv, const float* __restrict__ Wself,
    const float* __restrict__ Wneigh,
    _Float16* __restrict__ wqkvh, _Float16* __restrict__ wsageh, int L) {
  long total_q = (long)L * KQP * NQ;
  long total_s = (long)L * 128 * HID;
  for (long i = (long)blockIdx.x * blockDim.x + threadIdx.x;
       i < total_q + total_s; i += (long)gridDim.x * blockDim.x) {
    if (i < total_q) {
      int l = (int)(i / (KQP * NQ));
      int rem = (int)(i % (KQP * NQ));
      int kk = rem / NQ, nn = rem % NQ;
      float w = 0.f;
      if (kk < HID + PHID) {
        const float* W = (nn < HID) ? Wq : (nn < 2 * HID ? Wk : Wv);
        w = W[((long)l * (HID + PHID) + kk) * HID + (nn & (HID - 1))];
      }
      wqkvh[i] = (_Float16)w;
    } else {
      long j = i - total_q;
      int l = (int)(j / (128 * HID));
      int rem = (int)(j % (128 * HID));
      int kk = rem / HID, nn = rem % HID;
      float w = (kk < HID) ? Wself[((long)l * HID + kk) * HID + nn]
                           : Wneigh[((long)l * HID + (kk - HID)) * HID + nn];
      wsageh[j] = (_Float16)w;
    }
  }
}

// ------------------------------------------------- hc = tanh(cell_feat@Wc+bc)
__global__ __launch_bounds__(256) void encode_cells(
    const float* __restrict__ cf, const float* __restrict__ Wc,
    const float* __restrict__ bc, float* __restrict__ hc, int NC) {
  long i = (long)blockIdx.x * blockDim.x + threadIdx.x;
  if (i >= (long)NC * HID) return;
  int c = (int)(i / HID), j = (int)(i % HID);
  float s = bc[j];
  #pragma unroll
  for (int k2 = 0; k2 < CIN; ++k2) s += cf[(long)c * CIN + k2] * Wc[k2 * HID + j];
  hc[i] = tanhf(s);
}

// ------------------------------ hp = tanh(pin@Wp+bp); ew = tanh(hp@Wew+bew)
__global__ __launch_bounds__(256) void encode_pins(
    const float* __restrict__ pf, const float* __restrict__ Wp, const float* __restrict__ bp,
    const float* __restrict__ Wew, const float* __restrict__ bew,
    float* __restrict__ hp, float* __restrict__ ew, int E) {
  int e = blockIdx.x * blockDim.x + threadIdx.x;
  if (e >= E) return;
  float x[PINF];
  #pragma unroll
  for (int k2 = 0; k2 < PINF; ++k2) x[k2] = pf[(long)e * PINF + k2];
  float acc = bew[0];
  #pragma unroll
  for (int j = 0; j < PHID; ++j) {
    float s = bp[j];
    #pragma unroll
    for (int k2 = 0; k2 < PINF; ++k2) s += x[k2] * Wp[k2 * PHID + j];
    float h = tanhf(s);
    hp[(long)e * PHID + j] = h;
    acc += h * Wew[j];
  }
  ew[e] = tanhf(acc);
}

// ----------------------------------------------- deg[c] = #edges with dst=c
__global__ __launch_bounds__(256) void compute_deg(
    const int* __restrict__ ps, float* __restrict__ deg, int E) {
  int e = blockIdx.x * blockDim.x + threadIdx.x;
  if (e < E) atomicAdd(&deg[ps[e]], 1.f);
}

// -------------------------------------------------------------- QKV via WMMA
// x[E,80] = [hc[pins_src] | hp] ; q/k/v = x @ W[80,64] + b  (f16 in, f32 acc)
__global__ __launch_bounds__(256) void qkv_wmma(
    const float* __restrict__ hc, const float* __restrict__ hp,
    const int* __restrict__ ps, const _Float16* __restrict__ wh,
    const float* __restrict__ bq, const float* __restrict__ bk,
    const float* __restrict__ bv,
    float* __restrict__ qo, float* __restrict__ ko, float* __restrict__ vo,
    int E) {
  const int KQ = HID + PHID;  // 80
  __shared__ __align__(16) _Float16 sX[64][KQP];  // 12 KB
  __shared__ __align__(16) _Float16 sW[KQP][NQ];  // 36 KB
  int tid = threadIdx.x;
  long block_row = (long)blockIdx.x * 64;

  // stage weight tile: one TDM DMA (wave 0 only; TDM ignores EXEC)
#ifdef USE_TDM
  if (tid < 32) tdm_copy_to_lds(wh, &sW[0][0], KQP * NQ * 2);
#else
  {
    _Float16* d = &sW[0][0];
    for (int i = tid; i < KQP * NQ; i += 256) d[i] = wh[i];
  }
#endif
  // stage x tile (gather + f32->f16)
  for (int i = tid; i < 64 * KQP; i += 256) {
    int r = i / KQP, c = i % KQP;
    long e = block_row + r;
    float xv = 0.f;
    if (e < E) {
      if (c < HID)     xv = hc[(long)ps[e] * HID + c];
      else if (c < KQ) xv = hp[e * PHID + (c - HID)];
    }
    sX[r][c] = (_Float16)xv;
  }
  __syncthreads();

  int wave = tid >> 5, lane = tid & 31;
  int row_tile = wave >> 1;       // 0..3  (M tiles of 16)
  int n_half   = wave & 1;        // 6 N-tiles each
  int m = (row_tile << 4) + (lane & 15);
  int khalf = lane >> 4;

  v16h afr[3];                    // A fragments, ISA 16-bit 16x32 layout
  #pragma unroll
  for (int kt = 0; kt < 3; ++kt) {
    int kb = kt * 32 + khalf * 8;
    #pragma unroll
    for (int j = 0; j < 8; ++j) {
      int k0 = (j < 4 ? 2 * j : 16 + 2 * (j - 4)) + kb;
      afr[kt][2 * j]     = sX[m][k0];
      afr[kt][2 * j + 1] = sX[m][k0 + 1];
    }
  }
  #pragma unroll
  for (int nt = 0; nt < 6; ++nt) {
    int ncol = (n_half * 6 + nt) * 16 + (lane & 15);
    float bias;
    if (ncol < HID)          bias = bq[ncol];
    else if (ncol < 2 * HID) bias = bk[ncol - HID];
    else                     bias = bv[ncol - 2 * HID];
    v8f acc;
    #pragma unroll
    for (int i = 0; i < 8; ++i) acc[i] = bias;
    #pragma unroll
    for (int kt = 0; kt < 3; ++kt) {
      int kb = kt * 32 + khalf * 8;
      v16h bfr;
      #pragma unroll
      for (int j = 0; j < 8; ++j) {
        int k0 = (j < 4 ? 2 * j : 16 + 2 * (j - 4)) + kb;
        bfr[2 * j]     = sW[k0][ncol];
        bfr[2 * j + 1] = sW[k0 + 1][ncol];
      }
      acc = __builtin_amdgcn_wmma_f32_16x16x32_f16(
          false, afr[kt], false, bfr, (short)0, acc, false, false);
    }
    #pragma unroll
    for (int i = 0; i < 8; ++i) {
      long r = block_row + (row_tile << 4) + i + (khalf << 3);
      if (r < E) {
        float val = acc[i];
        if (ncol < HID)          qo[r * HID + ncol] = val;
        else if (ncol < 2 * HID) ko[r * HID + (ncol - HID)] = val;
        else                     vo[r * HID + (ncol - 2 * HID)] = val;
      }
    }
  }
}

// ------------------------- per-net mailbox attention, thread per (net, head)
__global__ __launch_bounds__(256) void attn_mailbox(
    const float* __restrict__ q, const float* __restrict__ k, const float* __restrict__ v,
    float* __restrict__ hn, int NN, int D) {
  int t = blockIdx.x * blockDim.x + threadIdx.x;
  if (t >= NN * NHEAD) return;
  int n = t / NHEAD, h = t % NHEAD;
  float acc[HD];
  #pragma unroll
  for (int i = 0; i < HD; ++i) acc[i] = 0.f;
  for (int d = 0; d < D; ++d) {
    const float* qr = q + ((long)(n * D + d)) * HID + h * HD;
    float qv[HD];
    #pragma unroll
    for (int i = 0; i < HD; ++i) qv[i] = qr[i] * 0.25f;  // scale = hd^-0.5
    float sc[DMAX], mx = -1e30f;
    for (int m2 = 0; m2 < D; ++m2) {
      const float* kr = k + ((long)(n * D + m2)) * HID + h * HD;
      float s = 0.f;
      #pragma unroll
      for (int i = 0; i < HD; ++i) s += qv[i] * kr[i];
      sc[m2] = s;
      mx = fmaxf(mx, s);
    }
    float se = 0.f;
    for (int m2 = 0; m2 < D; ++m2) { sc[m2] = __expf(sc[m2] - mx); se += sc[m2]; }
    float inv = 1.f / se;
    for (int m2 = 0; m2 < D; ++m2) {
      float w = sc[m2] * inv;
      const float* vr = v + ((long)(n * D + m2)) * HID + h * HD;
      #pragma unroll
      for (int i = 0; i < HD; ++i) acc[i] += w * vr[i];
    }
  }
  float invD = 1.f / (float)D;
  #pragma unroll
  for (int i = 0; i < HD; ++i) hn[(long)n * HID + h * HD + i] = acc[i] * invD;
}

// ------------------------------ scatter msg = hn[net]*ew into accum per cell
__global__ __launch_bounds__(256) void sage_aggregate(
    const float* __restrict__ hn, const float* __restrict__ ew,
    const int* __restrict__ ps, float* __restrict__ accum, int E, int D) {
  int e = blockIdx.x * blockDim.x + threadIdx.x;
  if (e >= E) return;
  int net = e / D;
  int c = ps[e];
  float w = ew[e];
  const float* h = hn + (long)net * HID;
  float* a = accum + (long)c * HID;
  #pragma unroll 8
  for (int j = 0; j < HID; ++j) atomicAdd(&a[j], h[j] * w);
}

// ------------- hc_out = [hc | accum/deg][NC,128] @ [Wself;Wneigh] + bsage
__global__ __launch_bounds__(256) void sage_wmma(
    const float* __restrict__ hc, const float* __restrict__ accum,
    const float* __restrict__ deg, const _Float16* __restrict__ wh,
    const float* __restrict__ bs, float* __restrict__ hc_out, int NC) {
  __shared__ __align__(16) _Float16 sX[64][128];   // 16 KB
  __shared__ __align__(16) _Float16 sW[128][HID];  // 16 KB
  int tid = threadIdx.x;
  long block_row = (long)blockIdx.x * 64;

#ifdef USE_TDM
  if (tid < 32) tdm_copy_to_lds(wh, &sW[0][0], 128 * HID * 2);
#else
  {
    _Float16* d = &sW[0][0];
    for (int i = tid; i < 128 * HID; i += 256) d[i] = wh[i];
  }
#endif
  for (int i = tid; i < 64 * 128; i += 256) {
    int r = i >> 7, c = i & 127;
    long row = block_row + r;
    float xv = 0.f;
    if (row < NC) {
      if (c < HID) xv = hc[row * HID + c];
      else {
        float dg = deg[row];
        if (dg < 1.f) dg = 1.f;
        xv = accum[row * HID + (c - HID)] / dg;
      }
    }
    sX[r][c] = (_Float16)xv;
  }
  __syncthreads();

  int wave = tid >> 5, lane = tid & 31;
  int row_tile = wave >> 1;
  int n_half   = wave & 1;   // 2 N-tiles each
  int m = (row_tile << 4) + (lane & 15);
  int khalf = lane >> 4;

  v16h afr[4];
  #pragma unroll
  for (int kt = 0; kt < 4; ++kt) {
    int kb = kt * 32 + khalf * 8;
    #pragma unroll
    for (int j = 0; j < 8; ++j) {
      int k0 = (j < 4 ? 2 * j : 16 + 2 * (j - 4)) + kb;
      afr[kt][2 * j]     = sX[m][k0];
      afr[kt][2 * j + 1] = sX[m][k0 + 1];
    }
  }
  #pragma unroll
  for (int nt = 0; nt < 2; ++nt) {
    int ncol = (n_half * 2 + nt) * 16 + (lane & 15);
    float bias = bs[ncol];
    v8f acc;
    #pragma unroll
    for (int i = 0; i < 8; ++i) acc[i] = bias;
    #pragma unroll
    for (int kt = 0; kt < 4; ++kt) {
      int kb = kt * 32 + khalf * 8;
      v16h bfr;
      #pragma unroll
      for (int j = 0; j < 8; ++j) {
        int k0 = (j < 4 ? 2 * j : 16 + 2 * (j - 4)) + kb;
        bfr[2 * j]     = sW[k0][ncol];
        bfr[2 * j + 1] = sW[k0 + 1][ncol];
      }
      acc = __builtin_amdgcn_wmma_f32_16x16x32_f16(
          false, afr[kt], false, bfr, (short)0, acc, false, false);
    }
    #pragma unroll
    for (int i = 0; i < 8; ++i) {
      long r = block_row + (row_tile << 4) + i + (khalf << 3);
      if (r < NC) hc_out[r * HID + ncol] = acc[i];
    }
  }
}

// ----------------------------------------------------- pairwise readout head
__global__ __launch_bounds__(256) void readout(
    const float* __restrict__ hc, const float* __restrict__ cs,
    const float* __restrict__ Wd, const float* __restrict__ bd,
    const float* __restrict__ Wa, const float* __restrict__ ba,
    const int* __restrict__ fa, const int* __restrict__ so,
    float* __restrict__ out, int EPT) {
  int t = blockIdx.x * blockDim.x + threadIdx.x;
  if (t >= EPT) return;
  int f = fa[t], s2 = so[t];
  float sd = bd[0], sa = ba[0];
  for (int j = 0; j < HID; ++j) {
    float a = hc[(long)f * HID + j];
    sd += a * Wd[j];
    sa += a * Wa[j];
  }
  for (int j = 0; j < HID; ++j) {
    float a = hc[(long)s2 * HID + j];
    sd += a * Wd[HID + j];
    sa += a * Wa[HID + j];
  }
  float dis = __expf(-2.f + 15.f * tanhf(sd));
  float ang = tanhf(sa) * 4.f;
  const float PI = 3.14159265358979323846f, eps = 1e-4f;
  float b0 = (cs[(long)f * 2 + 0] + cs[(long)s2 * 2 + 0]) * 0.5f;
  float b1 = (cs[(long)f * 2 + 1] + cs[(long)s2 * 2 + 1]) * 0.5f;
  float t0 = fabsf(b0 / (__cosf(ang * PI) + eps));
  float t1 = fabsf(b1 / (__sinf(ang * PI) + eps));
  out[t] = dis + fminf(t0, t1);
  out[(long)EPT + t] = ang;
}

// ---------------------------------------------------------------------- host
extern "C" void kernel_launch(void* const* d_in, const int* in_sizes, int n_in,
                              void* d_out, int out_size, void* d_ws, size_t ws_size,
                              hipStream_t stream) {
  const float* cell_feat = (const float*)d_in[0];
  const float* pin_feat  = (const float*)d_in[2];
  const float* cell_size = (const float*)d_in[3];
  const float* Wc  = (const float*)d_in[4];
  const float* bc  = (const float*)d_in[5];
  const float* Wp  = (const float*)d_in[8];
  const float* bp  = (const float*)d_in[9];
  const float* Wew = (const float*)d_in[10];
  const float* bew = (const float*)d_in[11];
  const float* Wq  = (const float*)d_in[12];
  const float* bq  = (const float*)d_in[13];
  const float* Wk  = (const float*)d_in[14];
  const float* bk  = (const float*)d_in[15];
  const float* Wv  = (const float*)d_in[16];
  const float* bv  = (const float*)d_in[17];
  const float* Wself  = (const float*)d_in[18];
  const float* Wneigh = (const float*)d_in[19];
  const float* bsage  = (const float*)d_in[20];
  const float* Wd  = (const float*)d_in[21];
  const float* bd  = (const float*)d_in[22];
  const float* Wa  = (const float*)d_in[23];
  const float* ba  = (const float*)d_in[24];
  const int* pins_src = (const int*)d_in[25];
  const int* fathers  = (const int*)d_in[26];
  const int* sons     = (const int*)d_in[27];

  int NC  = in_sizes[0] / CIN;
  int NN  = in_sizes[1] / 8;
  int E   = in_sizes[25];
  int EPT = in_sizes[26];
  int D   = E / NN;
  int L   = in_sizes[13] / HID;   // bq is [L, HID]
  const int KQ = HID + PHID;      // 80

  float* ws = (float*)d_ws;
  float* hcA   = ws; ws += (long)NC * HID;
  float* hcB   = ws; ws += (long)NC * HID;
  float* hp    = ws; ws += (long)E * PHID;
  float* ew    = ws; ws += E;
  float* deg   = ws; ws += NC + (NC & 1);          // keep 8B alignment
  float* qbuf  = ws; ws += (long)E * HID;
  float* kbuf  = ws; ws += (long)E * HID;
  float* vbuf  = ws; ws += (long)E * HID;
  float* hn    = ws; ws += (long)NN * HID;
  float* accum = ws; ws += (long)NC * HID;
  _Float16* wqkvh  = (_Float16*)ws;
  _Float16* wsageh = wqkvh + (long)L * KQP * NQ;
  (void)ws_size; (void)n_in; (void)out_size;

  // one-time prep: pack weights into f16 LDS-tile images
  {
    long total = (long)L * (KQP * NQ + 128 * HID);
    pack_weights_f16<<<dim3((unsigned)((total + 255) / 256)), 256, 0, stream>>>(
        Wq, Wk, Wv, Wself, Wneigh, wqkvh, wsageh, L);
  }
  {
    long n = (long)NC * HID;
    encode_cells<<<dim3((unsigned)((n + 255) / 256)), 256, 0, stream>>>(
        cell_feat, Wc, bc, hcA, NC);
  }
  encode_pins<<<dim3((unsigned)((E + 255) / 256)), 256, 0, stream>>>(
      pin_feat, Wp, bp, Wew, bew, hp, ew, E);
  zero_f32<<<dim3((unsigned)((NC + 255) / 256)), 256, 0, stream>>>(deg, NC);
  compute_deg<<<dim3((unsigned)((E + 255) / 256)), 256, 0, stream>>>(pins_src, deg, E);

  float* hc_cur = hcA;
  float* hc_nxt = hcB;
  for (int l = 0; l < L; ++l) {
    qkv_wmma<<<dim3((unsigned)((E + 63) / 64)), 256, 0, stream>>>(
        hc_cur, hp, pins_src,
        wqkvh + (long)l * KQP * NQ,
        bq + (long)l * HID, bk + (long)l * HID, bv + (long)l * HID,
        qbuf, kbuf, vbuf, E);
    attn_mailbox<<<dim3((unsigned)(((long)NN * NHEAD + 255) / 256)), 256, 0, stream>>>(
        qbuf, kbuf, vbuf, hn, NN, D);
    {
      long n = (long)NC * HID;
      zero_f32<<<dim3((unsigned)((n + 255) / 256)), 256, 0, stream>>>(accum, n);
    }
    sage_aggregate<<<dim3((unsigned)((E + 255) / 256)), 256, 0, stream>>>(
        hn, ew, pins_src, accum, E, D);
    sage_wmma<<<dim3((unsigned)((NC + 63) / 64)), 256, 0, stream>>>(
        hc_cur, accum, deg,
        wsageh + (long)l * 128 * HID,
        bsage + (long)l * HID, hc_nxt, NC);
    float* t = hc_cur; hc_cur = hc_nxt; hc_nxt = t;
  }

  readout<<<dim3((unsigned)((EPT + 255) / 256)), 256, 0, stream>>>(
      hc_cur, cell_size, Wd, bd, Wa, ba, fathers, sons, (float*)d_out, EPT);
}
